// GetContext_3891240370405
// MI455X (gfx1250) — compile-verified, matrix operand order checked by reading, not run on hardware
//
#include <hip/hip_runtime.h>
#include <cstddef>

#define NF 74
#define EF 12
#define GF 128

typedef __attribute__((ext_vector_type(16))) __bf16 v16bf;
typedef __attribute__((ext_vector_type(8)))  float  v8f;

// ---------------- bf16 helpers (round-to-nearest-even) ----------------
__device__ __forceinline__ unsigned short f2bf(float f) {
  unsigned u = __builtin_bit_cast(unsigned, f);
  return (unsigned short)((u + 0x7FFFu + ((u >> 16) & 1u)) >> 16);
}
__device__ __forceinline__ float bf2f(unsigned short h) {
  unsigned u = ((unsigned)h) << 16;
  return __builtin_bit_cast(float, u);
}

// Fragment element j (0..15) of a lane maps to K offset within a 32-wide step:
//   koff = (lane>>4)*8;  K = 32*t + koff + (j<8 ? j : 8+j)
__device__ __forceinline__ int frag_k(int lane, int t, int j) {
  return 32 * t + ((lane >> 4) << 3) + (j < 8 ? j : 8 + j);
}

// ---------------- A-matrix accessors ----------------
struct AccF32 {               // dense fp32 [M][ld]
  const float* p; int ld;
  __device__ float get(int r, int c) const { return p[(size_t)r * ld + c]; }
};
struct AccEdgeConcat {        // cat(node_feats[src], edge_feats) -> [E][86]
  const float* nf; const float* ef; const int* src;
  __device__ float get(int e, int c) const {
    return (c < NF) ? nf[(size_t)src[e] * NF + c] : ef[(size_t)e * EF + (c - NF)];
  }
};
struct AccBf16 {              // bf16-bits matrix [M][ld]
  const unsigned short* p; int ld;
  __device__ float get(int r, int c) const { return bf2f(p[(size_t)r * ld + c]); }
};
struct AccConcat3 {           // cat over 3 heads of [heads][M][128]; optional elu
  const float* base; int Mrows; int applyElu;
  __device__ float get(int r, int c) const {
    int h = c >> 7, cc = c & 127;
    float v = base[((size_t)h * Mrows + r) * GF + cc];
    if (applyElu) v = v > 0.f ? v : (__expf(v) - 1.f);
    return v;
  }
};

// ---------------- epilogues ----------------
struct EpiStoreF32 {          // out fp32, optional leaky
  float* out; int ld; int leaky;
  __device__ void operator()(int r, int c, float v) const {
    if (leaky) v = v > 0.f ? v : 0.01f * v;
    out[(size_t)r * ld + c] = v;
  }
};
struct EpiStoreBf16Leaky {    // he1k: leaky then store bf16 bits
  unsigned short* out; int ld;
  __device__ void operator()(int r, int c, float v) const {
    v = v > 0.f ? v : 0.01f * v;
    out[(size_t)r * ld + c] = f2bf(v);
  }
};
struct EpiEtScatter {         // msg = alpha * (he1k@tw + tb); segment_sum into ctx
  const float* ew;            // exp(lg - m[dst])  [E] (head base applied)
  const float* ssum;          // segment sums      [N]
  const int*   dst;
  float*       ctx;           // [N][128] accumulator (head base applied)
  __device__ void operator()(int e, int c, float v) const {
    int d = dst[e];
    float alpha = ew[e] / ssum[d];
    atomicAdd(ctx + (size_t)d * GF + c, v * alpha);
  }
};

// ---------------- generic tall-skinny WMMA GEMM ----------------
// Block = 256 threads (8 wave32s). Tile: 16 rows x 128 cols (blockIdx.y picks
// the 128-col block when Ncols>128). K processed in 128-wide chunks.
// LDS holds A and W tiles PRE-SWIZZLED into WMMA fragment-major order so each
// fragment load is 2 contiguous ds_load_b128 per matrix, conflict-free.
template <class AAcc, class Epi>
__global__ __launch_bounds__(256)
void wmma_gemm(AAcc A, const float* __restrict__ W, int ldW,
               const float* __restrict__ bias, int M, int K, Epi epi) {
  // fragment-major: [k-step t][lane][16 bf16]  (A) ; [ntile w][t][lane][16] (W)
  __shared__ __align__(16) unsigned short sAf[4][32][16];      //  4 KB
  __shared__ __align__(16) unsigned short sWf[8][4][32][16];   // 32 KB

  const int tid  = threadIdx.x;
  const int wave = tid >> 5;
  const int lane = tid & 31;
  const int row0 = blockIdx.x * 16;
  const int col0 = blockIdx.y * 128;

  v8f acc = {};
  for (int k0 = 0; k0 < K; k0 += 128) {
    // ---- stage A fragments: 4*32*16 = 2048 elems (8 per thread) ----
    for (int i = tid; i < 4 * 32 * 16; i += 256) {
      int j = i & 15, l = (i >> 4) & 31, t = i >> 9;
      int row = row0 + (l & 15);
      int kk  = k0 + frag_k(l, t, j);
      float v = 0.f;
      if (row < M && kk < K) v = A.get(row, kk);
      sAf[t][l][j] = f2bf(v);
    }
    // ---- stage W fragments: 8*4*32*16 = 16384 elems (64 per thread) ----
    for (int i = tid; i < 8 * 4 * 32 * 16; i += 256) {
      int j = i & 15, l = (i >> 4) & 31, t = (i >> 9) & 3, w = i >> 11;
      int kk   = k0 + frag_k(l, t, j);
      int colg = col0 + (w << 4) + (l & 15);
      float v = 0.f;
      if (kk < K) v = W[(size_t)kk * ldW + colg];
      sWf[w][t][l][j] = f2bf(v);
    }
    // prefetch next weight chunk while we compute (global_prefetch_b8 path)
    if (k0 + 128 < K) {
      int kk = k0 + 128 + (tid >> 1);
      if (kk < K) __builtin_prefetch(W + (size_t)kk * ldW + col0 + (tid & 1) * 64, 0, 3);
    }
    __syncthreads();

    int rem  = K - k0;
    int tmax = (rem >= 128 ? 128 : ((rem + 31) & ~31)) >> 5;
    const uint4* pa = (const uint4*)&sAf[0][lane][0];
    const uint4* pb = (const uint4*)&sWf[wave][0][lane][0];
    for (int t = 0; t < tmax; ++t) {
      union { v16bf v; uint4 q[2]; } a, b;
      a.q[0] = pa[t * 64 + 0];           // [t][lane] slab: 32 lanes * 32B = 1024B = 64 uint4
      a.q[1] = pa[t * 64 + 1];
      b.q[0] = pb[t * 64 + 0];
      b.q[1] = pb[t * 64 + 1];
      acc = __builtin_amdgcn_wmma_f32_16x16x32_bf16(
          /*neg_a=*/false, a.v, /*neg_b=*/false, b.v,
          /*c_mod=*/(short)0, acc, /*reuse_a=*/false, /*reuse_b=*/false);
    }
    __syncthreads();
  }

  // C/D layout: VGPR r -> row r (lanes 0-15) / row r+8 (lanes 16-31); col = lane&15
  const int rbase = (lane >> 4) * 8;
  const int col   = col0 + wave * 16 + (lane & 15);
  const float bj  = bias ? bias[col] : 0.f;
#pragma unroll
  for (int r = 0; r < 8; ++r) {
    int grow = row0 + rbase + r;
    if (grow < M) epi(grow, col, acc[r] + bj);
  }
}

// ---------------- edge logits: lg = leaky(cat(hv[dst], he1k) @ w2 + b2) ----------------
__global__ __launch_bounds__(256)
void lg_kernel(const float* __restrict__ hv,            // [3][N][128]
               const unsigned short* __restrict__ he1k, // [3][E][128] bf16
               const float* w2_0, const float* b2_0,
               const float* w2_1, const float* b2_1,
               const float* w2_2, const float* b2_2,
               const int* __restrict__ dst,
               float* __restrict__ lg,                   // [3][E]
               unsigned int* __restrict__ menc,          // [3][N] ordered-uint max
               int E_, int N_) {
  int gw = blockIdx.x * 8 + (threadIdx.x >> 5);
  int lane = threadIdx.x & 31;
  if (gw >= 3 * E_) return;
  int head = gw / E_;
  int e    = gw - head * E_;
  const float* w2 = head == 0 ? w2_0 : head == 1 ? w2_1 : w2_2;
  const float* b2 = head == 0 ? b2_0 : head == 1 ? b2_1 : b2_2;
  int d = dst[e];
  const float* hvrow = hv + ((size_t)head * N_ + d) * GF;
  const unsigned short* herow = he1k + ((size_t)head * E_ + e) * GF;
  float s = 0.f;
  for (int c = lane; c < GF; c += 32) {
    s += hvrow[c] * w2[c];
    s += bf2f(herow[c]) * w2[GF + c];
  }
  for (int off = 16; off; off >>= 1) s += __shfl_down(s, off, 32);
  if (lane == 0) {
    float v = s + b2[0];
    v = v > 0.f ? v : 0.01f * v;
    lg[(size_t)head * E_ + e] = v;
    unsigned bits = __builtin_bit_cast(unsigned, v);
    unsigned enc  = (bits & 0x80000000u) ? ~bits : (bits | 0x80000000u);
    atomicMax(menc + (size_t)head * N_ + d, enc);
  }
}

// ---------------- softmax numerator + denominator ----------------
__global__ void exp_kernel(const float* __restrict__ lg,
                           const unsigned int* __restrict__ menc,
                           const int* __restrict__ dst,
                           float* __restrict__ ew, float* __restrict__ ssum,
                           int E_, int N_) {
  int i = blockIdx.x * blockDim.x + threadIdx.x;
  if (i >= 3 * E_) return;
  int head = i / E_, e = i - head * E_;
  int d = dst[e];
  unsigned enc = menc[(size_t)head * N_ + d];
  unsigned bits = (enc & 0x80000000u) ? (enc & 0x7FFFFFFFu) : ~enc;
  float m = __builtin_bit_cast(float, bits);
  float v = __expf(lg[i] - m);
  ew[i] = v;
  atomicAdd(ssum + (size_t)head * N_ + d, v);
}

// ---------------- GRU gates + output relu ----------------
__global__ void gru_kernel(const float* __restrict__ gi, const float* __restrict__ gh,
                           const float* __restrict__ h, float* __restrict__ out, int N_) {
  int i = blockIdx.x * blockDim.x + threadIdx.x;
  if (i >= N_ * GF) return;
  int n = i >> 7, c = i & 127;
  const float* gin = gi + (size_t)n * 3 * GF;
  const float* ghn = gh + (size_t)n * 3 * GF;
  float ir = gin[c], iz = gin[GF + c], inn = gin[2 * GF + c];
  float hr = ghn[c], hz = ghn[GF + c], hn = ghn[2 * GF + c];
  float r = 1.f / (1.f + __expf(-(ir + hr)));
  float z = 1.f / (1.f + __expf(-(iz + hz)));
  float nn = tanhf(inn + r * hn);
  float hv = h[i];
  float o  = (1.f - z) * nn + z * hv;
  out[i] = o > 0.f ? o : 0.f;
}

// ---------------- host launcher ----------------
extern "C" void kernel_launch(void* const* d_in, const int* in_sizes, int n_in,
                              void* d_out, int out_size, void* d_ws, size_t ws_size,
                              hipStream_t stream) {
  const int N = in_sizes[0] / NF;
  const int E = in_sizes[1] / EF;

  const float* node_feats = (const float*)d_in[0];
  const float* edge_feats = (const float*)d_in[1];
  const float *pn_w[3], *pn_b[3], *pe1_w[3], *pe1_b[3], *pe2_w[3], *pe2_b[3],
              *et_w[3], *et_b[3];
  for (int k = 0; k < 3; ++k) {
    pn_w[k]  = (const float*)d_in[2 + 8 * k];
    pn_b[k]  = (const float*)d_in[3 + 8 * k];
    pe1_w[k] = (const float*)d_in[4 + 8 * k];
    pe1_b[k] = (const float*)d_in[5 + 8 * k];
    pe2_w[k] = (const float*)d_in[6 + 8 * k];
    pe2_b[k] = (const float*)d_in[7 + 8 * k];
    et_w[k]  = (const float*)d_in[8 + 8 * k];
    et_b[k]  = (const float*)d_in[9 + 8 * k];
  }
  const float* mca_w   = (const float*)d_in[26];
  const float* mca_b   = (const float*)d_in[27];
  const float* mcn_w   = (const float*)d_in[28];
  const float* mcn_b   = (const float*)d_in[29];
  const float* gru_wih = (const float*)d_in[30];
  const float* gru_whh = (const float*)d_in[31];
  const float* gru_bih = (const float*)d_in[32];
  const float* gru_bhh = (const float*)d_in[33];
  const int*   src     = (const int*)d_in[34];
  const int*   dst     = (const int*)d_in[35];

  // workspace carve-out
  char* ws = (char*)d_ws;
  size_t off = 0;
  auto carve = [&](size_t bytes) -> char* {
    char* p = ws + off;
    off = (off + bytes + 255) & ~(size_t)255;
    return p;
  };
  float*          hv     = (float*)carve((size_t)3 * N * GF * 4);          // projected nodes
  unsigned short* he1k   = (unsigned short*)carve((size_t)3 * E * GF * 2); // edge proj, bf16
  float*          lg     = (float*)carve((size_t)3 * E * 4);
  float*          ew     = (float*)carve((size_t)3 * E * 4);
  unsigned int*   menc   = (unsigned int*)carve((size_t)3 * N * 4);
  float*          ssum   = (float*)carve((size_t)3 * N * 4);
  float*          ctx    = (float*)carve((size_t)3 * N * GF * 4);
  float*          cctx   = (float*)carve((size_t)N * GF * 4);              // context
  float*          hbuf   = (float*)carve((size_t)N * GF * 4);              // h
  float*          gi     = (float*)carve((size_t)N * 3 * GF * 4);
  float*          gh     = (float*)carve((size_t)N * 3 * GF * 4);
  (void)ws_size; (void)n_in;

  // zero the accumulators (graph-capture-safe)
  hipMemsetAsync(menc, 0, (size_t)3 * N * 4, stream);
  hipMemsetAsync(ssum, 0, (size_t)3 * N * 4, stream);
  hipMemsetAsync(ctx,  0, (size_t)3 * N * GF * 4, stream);

  const dim3 blk(256);
  const int gN = (N + 15) / 16;
  const int gE = (E + 15) / 16;

  // 1) hv[k] = leaky(node_feats @ pn_w + pn_b)   [N,74]x[74,128]
  for (int k = 0; k < 3; ++k) {
    AccF32 a{node_feats, NF};
    EpiStoreF32 e{hv + (size_t)k * N * GF, GF, 1};
    wmma_gemm<<<dim3(gN, 1), blk, 0, stream>>>(a, pn_w[k], GF, pn_b[k], N, NF, e);
  }
  // 2) he1k[k] = leaky(cat(nf[src], ef) @ pe1_w + pe1_b)  [E,86]x[86,128], bf16 out
  for (int k = 0; k < 3; ++k) {
    AccEdgeConcat a{node_feats, edge_feats, src};
    EpiStoreBf16Leaky e{he1k + (size_t)k * E * GF, GF};
    wmma_gemm<<<dim3(gE, 1), blk, 0, stream>>>(a, pe1_w[k], GF, pe1_b[k], E, NF + EF, e);
  }
  // 3) edge logits + running segment max
  {
    int waves = 3 * E;
    lg_kernel<<<dim3((waves + 7) / 8), blk, 0, stream>>>(
        hv, he1k, pe2_w[0], pe2_b[0], pe2_w[1], pe2_b[1], pe2_w[2], pe2_b[2],
        dst, lg, menc, E, N);
  }
  // 4) softmax numerator + denominator
  exp_kernel<<<dim3((3 * E + 255) / 256), blk, 0, stream>>>(lg, menc, dst, ew, ssum, E, N);
  // 5) msg GEMM fused with alpha-scale + segment_sum scatter into ctx
  for (int k = 0; k < 3; ++k) {
    AccBf16 a{he1k + (size_t)k * E * GF, GF};
    EpiEtScatter e{ew + (size_t)k * E, ssum + (size_t)k * N, dst,
                   ctx + (size_t)k * N * GF};
    wmma_gemm<<<dim3(gE, 1), blk, 0, stream>>>(a, et_w[k], GF, et_b[k], E, GF, e);
  }
  // 6) context = elu(ctx) concat @ mca_w + mca_b   [N,384]x[384,128]
  {
    AccConcat3 a{ctx, N, 1};
    EpiStoreF32 e{cctx, GF, 0};
    wmma_gemm<<<dim3(gN, 1), blk, 0, stream>>>(a, mca_w, GF, mca_b, N, 3 * GF, e);
  }
  // 7) h = hv concat @ mcn_w + mcn_b               [N,384]x[384,128]
  {
    AccConcat3 a{hv, N, 0};
    EpiStoreF32 e{hbuf, GF, 0};
    wmma_gemm<<<dim3(gN, 1), blk, 0, stream>>>(a, mcn_w, GF, mcn_b, N, 3 * GF, e);
  }
  // 8) gi = context @ gru_wih + bih; gh = h @ gru_whh + bhh   [N,128]x[128,384]
  {
    AccF32 a{cctx, GF};
    EpiStoreF32 e{gi, 3 * GF, 0};
    wmma_gemm<<<dim3(gN, 3), blk, 0, stream>>>(a, gru_wih, 3 * GF, gru_bih, N, GF, e);
  }
  {
    AccF32 a{hbuf, GF};
    EpiStoreF32 e{gh, 3 * GF, 0};
    wmma_gemm<<<dim3(gN, 3), blk, 0, stream>>>(a, gru_whh, 3 * GF, gru_bhh, N, GF, e);
  }
  // 9) GRU gates + relu -> d_out [N,128]
  gru_kernel<<<dim3((N * GF + 255) / 256), blk, 0, stream>>>(gi, gh, hbuf, (float*)d_out, N);
}
